// Recurrence_17927193494257
// MI455X (gfx1250) — matrix operand order repaired and checked by reading, not compiled, not run on hardware
//
#include <hip/hip_runtime.h>
#include <math.h>

typedef _Float16 h16;
typedef __attribute__((ext_vector_type(16))) _Float16 v16h;
typedef __attribute__((ext_vector_type(8)))  _Float16 v8h;
typedef __attribute__((ext_vector_type(8)))  float    v8f;

#define T_   64
#define N_   4096
#define H_   64
#define L_   128
#define OBS_ 64
#define D_   194
#define TOT_ 72

static __device__ inline v8f zero8() {
  v8f z;
#pragma unroll
  for (int j = 0; j < 8; ++j) z[j] = 0.f;
  return z;
}

static __device__ inline v8f wmma_f16(v16h a, v16h b, v8f c) {
  // D = A(16x32 f16) * B(32x16 f16) + C(16x16 f32)
  return __builtin_amdgcn_wmma_f32_16x16x32_f16(false, a, false, b, (short)0, c,
                                                false, false);
}

// A-fragment (16x32): lane holds row (lane&15); K = half*16 + j.
// `row` points at this lane's row with the 32-wide K-chunk base already applied.
static __device__ inline v16h fragA_h16(const h16* row, int half) {
  const v8h lo = *(const v8h*)(row + half * 16);
  const v8h hi = *(const v8h*)(row + half * 16 + 8);
  v16h f;
#pragma unroll
  for (int j = 0; j < 8; ++j) { f[j] = lo[j]; f[8 + j] = hi[j]; }
  return f;
}

static __device__ inline v16h fragA_f32(const float* row, int half) {
  v16h f;
#pragma unroll
  for (int j = 0; j < 8; ++j) {
    f[j]     = (h16)row[half * 16 + j];
    f[8 + j] = (h16)row[half * 16 + 8 + j];
  }
  return f;
}

// B-fragment (32x16): lane holds column (lane&15); K = half*16 + j.
// `p` points at W[col*din + kbase + half*16] (B = W^T, W row-major [dout][din]).
static __device__ inline v16h fragB_any(const h16* p) {
  const v8h lo = *(const v8h*)(p);
  const v8h hi = *(const v8h*)(p + 8);
  v16h f;
#pragma unroll
  for (int j = 0; j < 8; ++j) { f[j] = lo[j]; f[8 + j] = hi[j]; }
  return f;
}

static __device__ inline v16h fragB_any(const float* p) {
  v16h f;
#pragma unroll
  for (int j = 0; j < 16; ++j) f[j] = (h16)p[j];
  return f;
}

// ---------------------------------------------------------------------------
// Phase 1: bidirectional GRU over L=128 steps. One wave owns 16 batch rows.
// grid = (N/128, 2 directions), block = 256 (8 waves).
// Writes M[l][n][dir*64 + k] as f16 into the workspace.
// ---------------------------------------------------------------------------
__global__ void __launch_bounds__(256)
gru_kernel(const float* __restrict__ inputs,
           const float* __restrict__ fWih, const float* __restrict__ fWhh,
           const float* __restrict__ fBih, const float* __restrict__ fBhh,
           const float* __restrict__ bWih, const float* __restrict__ bWhh,
           const float* __restrict__ bBih, const float* __restrict__ bBhh,
           h16* __restrict__ M)
{
  __shared__ __align__(16) h16 sWhh[192 * 64];      // 24 KB (f16)
  __shared__ float sWih[192], sBih[192], sBhh[192]; // 2.25 KB
  __shared__ float sXt[8][16];                      // per-wave x tile
  __shared__ __align__(16) h16 sH[8][16 * 64];      // per-wave h, 16 KB total

  const int tid  = threadIdx.x;
  const int w    = tid >> 5, lane = tid & 31;
  const int ncol = lane & 15, half = lane >> 4;
  const int dir  = blockIdx.y;
  const int rowBase = blockIdx.x * 128 + w * 16;

  const float* Wih = dir ? bWih : fWih;
  const float* Whh = dir ? bWhh : fWhh;
  const float* Bih = dir ? bBih : fBih;
  const float* Bhh = dir ? bBhh : fBhh;

  for (int i = tid; i < 192 * 64; i += 256) sWhh[i] = (h16)Whh[i];
  for (int i = tid; i < 192; i += 256) {
    sWih[i] = Wih[i]; sBih[i] = Bih[i]; sBhh[i] = Bhh[i];
  }
  for (int i = lane; i < 16 * 64; i += 32) sH[w][i] = (h16)0.f;
  __syncthreads();

  v16h ha0, ha1;   // h as A-fragments, K-chunks [0..31], [32..63]
#pragma unroll
  for (int j = 0; j < 16; ++j) { ha0[j] = (h16)0.f; ha1[j] = (h16)0.f; }

  for (int s = 0; s < L_; ++s) {
    const int lf = dir ? (L_ - 1 - s) : s;

    // x[m] = inputs[0, rowBase+m, OBS+lf]
    if (lane < 16)
      sXt[w][lane] = inputs[(size_t)(rowBase + lane) * D_ + OBS_ + lf];
    __syncthreads();

    // gh = h @ Whh^T : 12 output tiles (j = gate index 0..191), K = 64
    v8f acc[12];
#pragma unroll
    for (int c = 0; c < 12; ++c) {
      const h16* wr = &sWhh[(c * 16 + ncol) * 64 + half * 16];
      v8f z = zero8();
      z = wmma_f16(ha0, fragB_any(wr), z);
      z = wmma_f16(ha1, fragB_any(wr + 32), z);
      acc[c] = z;
    }

    // Gate math. Tiles c, c+4, c+8 give r/z/n for the SAME (lane, v) mapping.
#pragma unroll
    for (int c = 0; c < 4; ++c) {
      const int k = c * 16 + ncol;
#pragma unroll
      for (int v = 0; v < 8; ++v) {
        const int m = v + 8 * half;
        const float x   = sXt[w][m];
        const float gR  = sWih[k] * x       + sBih[k]       + acc[c][v]     + sBhh[k];
        const float gZ  = sWih[k + 64] * x  + sBih[k + 64]  + acc[c + 4][v] + sBhh[k + 64];
        const float giN = sWih[k + 128] * x + sBih[k + 128];
        const float ghN = acc[c + 8][v] + sBhh[k + 128];
        const float r  = 1.f / (1.f + __expf(-gR));
        const float z  = 1.f / (1.f + __expf(-gZ));
        const float ng = tanhf(giN + r * ghN);
        const float ho = (float)sH[w][m * 64 + k];
        sH[w][m * 64 + k] = (h16)((1.f - z) * ng + z * ho);
      }
    }
    __syncthreads();

    // Store this step's hidden tile to M[lf, rowBase.., dir*64..] (64B/lane).
    {
      const int r2 = lane >> 1, hh = lane & 1;
      const v8h* src = (const v8h*)&sH[w][r2 * 64 + hh * 32];
      v8h* dst = (v8h*)(M + ((size_t)lf * N_ + rowBase + r2) * 128 + dir * 64 + hh * 32);
      dst[0] = src[0]; dst[1] = src[1]; dst[2] = src[2]; dst[3] = src[3];
    }

    // Reload h as A-fragments for the next step (D->A transpose via LDS).
    const h16* hr = &sH[w][ncol * 64];
    ha0 = fragA_h16(hr, half);
    ha1 = fragA_h16(hr + 32, half);
  }
}

// ---------------------------------------------------------------------------
// Phase 2: T-step scan with data-dependent gather r = M[P]. One wave owns
// 16 batch rows for all T. grid = N/128 blocks, block = 256 (8 waves).
// ---------------------------------------------------------------------------
struct P2Args {
  const float *inputs, *rnn, *anoise, *bunif;
  const float *a0W, *a0b, *a1W, *a1b, *amW, *amb, *logstd;
  const float *c0W, *c0b, *c1W, *c1b, *coW, *cob;
  const float *b0W, *b0b, *b1W, *b1b, *boW, *bob;
  const h16* M;
  float* out;
};

// Two-layer trunk: t1 = A@W0^T + b0 (linear), t2 = relu(t1@W1^T + b1).
// W0 may be LDS f16 or global f32; W1 is global f32. Result left in D-layout.
template <int NC, typename WT>
static __device__ inline void trunk_eval(const v16h* a, const WT* W0, int din0,
                                         const float* b0, const float* W1,
                                         const float* b1, h16* tbuf,
                                         int ncol, int half, v8f acc2[4])
{
  v8f acc1[4];
#pragma unroll
  for (int c = 0; c < 4; ++c) {
    v8f z = zero8();
#pragma unroll
    for (int k = 0; k < NC; ++k)
      z = wmma_f16(a[k], fragB_any(&W0[(c * 16 + ncol) * din0 + k * 32 + half * 16]), z);
    acc1[c] = z;
  }
#pragma unroll
  for (int c = 0; c < 4; ++c) {
    const float bb = b0[c * 16 + ncol];
#pragma unroll
    for (int v = 0; v < 8; ++v)
      tbuf[(v + 8 * half) * 64 + c * 16 + ncol] = (h16)(acc1[c][v] + bb);
  }
  __syncthreads();
  const v16h t0 = fragA_h16(&tbuf[ncol * 64], half);
  const v16h t1 = fragA_h16(&tbuf[ncol * 64 + 32], half);
#pragma unroll
  for (int c = 0; c < 4; ++c) {
    v8f z = zero8();
    z = wmma_f16(t0, fragB_any(&W1[(c * 16 + ncol) * 64 + half * 16]), z);
    z = wmma_f16(t1, fragB_any(&W1[(c * 16 + ncol) * 64 + 32 + half * 16]), z);
    const float bb = b1[c * 16 + ncol];
#pragma unroll
    for (int v = 0; v < 8; ++v) z[v] = fmaxf(z[v] + bb, 0.f);
    acc2[c] = z;
  }
  __syncthreads();
}

// res[v] = dot(t2[row v+8*half][:], wout) via xor-shuffle reduction over 16 lanes.
static __device__ inline void dot_out(const v8f acc2[4], const float wv[4],
                                      float res[8]) {
#pragma unroll
  for (int v = 0; v < 8; ++v) {
    float p = acc2[0][v] * wv[0] + acc2[1][v] * wv[1] +
              acc2[2][v] * wv[2] + acc2[3][v] * wv[3];
#pragma unroll
    for (int off = 1; off < 16; off <<= 1) p += __shfl_xor(p, off);
    res[v] = p;
  }
}

__global__ void __launch_bounds__(256)
scan_kernel(P2Args A)
{
  __shared__ __align__(16) h16 sA0[64 * 128];  // actor0.W  (f16, 16 KB)
  __shared__ __align__(16) h16 sC0[64 * 128];  // critic0.W (f16, 16 KB)
  __shared__ __align__(16) h16 sT[8][16 * 64]; // per-wave transpose buf (16 KB)
  __shared__ int sP[8][16];                    // per-row position pointer

  const int tid  = threadIdx.x;
  const int w    = tid >> 5, lane = tid & 31;
  const int ncol = lane & 15, half = lane >> 4;
  const int rowBase = blockIdx.x * 128 + w * 16;

  for (int i = tid; i < 64 * 128; i += 256) {
    sA0[i] = (h16)A.a0W[i];
    sC0[i] = (h16)A.c0W[i];
  }
  if (lane < 16)
    sP[w][lane] = ((int)A.rnn[(size_t)(rowBase + lane) * TOT_ + (TOT_ - 1)]) & (L_ - 1);
  __syncthreads();

  const float scaleV = __expf(A.logstd[0]);

  for (int t = 0; t < T_; ++t) {
    // Gather r = M[P[m], n, :] as 4 A-fragments (16B aligned f16 loads).
    const int pm = sP[w][ncol];
    const h16* rrow = A.M + ((size_t)pm * N_ + (rowBase + ncol)) * 128;
    v16h ra[4];
#pragma unroll
    for (int cc = 0; cc < 4; ++cc) ra[cc] = fragA_h16(rrow + cc * 32, half);

    // beta input = concat(obs_t, r): chunks 0..1 obs, 2..5 r.
    const float* orow = A.inputs + ((size_t)t * N_ + rowBase + ncol) * D_;
    v16h bch[6];
    bch[0] = fragA_f32(orow, half);
    bch[1] = fragA_f32(orow + 32, half);
    bch[2] = ra[0]; bch[3] = ra[1]; bch[4] = ra[2]; bch[5] = ra[3];

    v8f acc2[4];
    float locs[8], vs[8], l0s[8], l1s[8];

    trunk_eval<4>(ra, sA0, 128, A.a0b, A.a1W, A.a1b, sT[w], ncol, half, acc2);
    {
      const float wv[4] = {A.amW[ncol], A.amW[16 + ncol], A.amW[32 + ncol], A.amW[48 + ncol]};
      dot_out(acc2, wv, locs);
    }

    trunk_eval<4>(ra, sC0, 128, A.c0b, A.c1W, A.c1b, sT[w], ncol, half, acc2);
    {
      const float wv[4] = {A.coW[ncol], A.coW[16 + ncol], A.coW[32 + ncol], A.coW[48 + ncol]};
      dot_out(acc2, wv, vs);
    }

    trunk_eval<6>(bch, A.b0W, 192, A.b0b, A.b1W, A.b1b, sT[w], ncol, half, acc2);
    {
      const float wv0[4] = {A.boW[ncol], A.boW[16 + ncol], A.boW[32 + ncol], A.boW[48 + ncol]};
      const float wv1[4] = {A.boW[64 + ncol], A.boW[80 + ncol], A.boW[96 + ncol], A.boW[112 + ncol]};
      dot_out(acc2, wv0, l0s);
      dot_out(acc2, wv1, l1s);
    }

    // Scalar tail: lanes 0 & 16 handle 8 rows each.
    if (ncol == 0) {
      const float am_b = A.amb[0], co_b = A.cob[0];
      const float bo0 = A.bob[0], bo1 = A.bob[1];
#pragma unroll 1
      for (int v = 0; v < 8; ++v) {
        const int m = v + 8 * half, n = rowBase + m;
        const size_t ib = (size_t)t * N_ + n;
        const float At  = A.inputs[ib * D_ + (D_ - 2)];
        const float Btf = A.inputs[ib * D_ + (D_ - 1)];
        const float eps = A.anoise[ib], u = A.bunif[ib];
        const float loc = locs[v] + am_b;
        const float vv  = vs[v] + co_b;
        const float l0 = l0s[v] + bo0, l1 = l1s[v] + bo1;
        const float p0 = 1.f / (1.f + __expf(l1 - l0)); // softmax-2
        const float a_samp = loc + scaleV * eps;
        const float Aout = (At < 0.f) ? a_samp : At;
        const int Bti  = (int)Btf;
        const int Bout = (Bti < 0) ? ((u >= p0) ? 1 : 0) : Bti;
        const int Pn   = (sP[w][m] + Bout) & (L_ - 1);
        sP[w][m] = Pn;
        float* o = A.out + ib * TOT_;
        o[0] = Aout;           o[1] = (float)Bout;
        o[2] = loc;            o[3] = scaleV;
        o[4] = p0;             o[5] = 1.f - p0;
        o[6] = vv;             o[TOT_ - 1] = (float)Pn;
      }
    }
    __syncthreads();

    // Broadcast h_state (rnn_hxs[:,7:71]) into out[...,7:71].
    for (int idx = lane; idx < 16 * 64; idx += 32) {
      const int m = idx >> 6, k = idx & 63, n = rowBase + m;
      A.out[((size_t)t * N_ + n) * TOT_ + 7 + k] = A.rnn[(size_t)n * TOT_ + 7 + k];
    }
  }
}

// ---------------------------------------------------------------------------
extern "C" void kernel_launch(void* const* d_in, const int* in_sizes, int n_in,
                              void* d_out, int out_size, void* d_ws, size_t ws_size,
                              hipStream_t stream)
{
  (void)in_sizes; (void)n_in; (void)out_size; (void)ws_size;

  const float* inputs = (const float*)d_in[0];
  const float* rnn    = (const float*)d_in[1];
  const float* fWih = (const float*)d_in[2], *fWhh = (const float*)d_in[3];
  const float* fBih = (const float*)d_in[4], *fBhh = (const float*)d_in[5];
  const float* bWih = (const float*)d_in[6], *bWhh = (const float*)d_in[7];
  const float* bBih = (const float*)d_in[8], *bBhh = (const float*)d_in[9];

  h16* M = (h16*)d_ws; // (L, N, 2H) f16 = 128 MB

  gru_kernel<<<dim3(N_ / 128, 2), 256, 0, stream>>>(
      inputs, fWih, fWhh, fBih, fBhh, bWih, bWhh, bBih, bBhh, M);

  P2Args A;
  A.inputs = inputs; A.rnn = rnn;
  A.a0W = (const float*)d_in[10]; A.a0b = (const float*)d_in[11];
  A.a1W = (const float*)d_in[12]; A.a1b = (const float*)d_in[13];
  A.amW = (const float*)d_in[14]; A.amb = (const float*)d_in[15];
  A.logstd = (const float*)d_in[16];
  A.c0W = (const float*)d_in[17]; A.c0b = (const float*)d_in[18];
  A.c1W = (const float*)d_in[19]; A.c1b = (const float*)d_in[20];
  A.coW = (const float*)d_in[21]; A.cob = (const float*)d_in[22];
  A.b0W = (const float*)d_in[23]; A.b0b = (const float*)d_in[24];
  A.b1W = (const float*)d_in[25]; A.b1b = (const float*)d_in[26];
  A.boW = (const float*)d_in[27]; A.bob = (const float*)d_in[28];
  A.anoise = (const float*)d_in[29]; A.bunif = (const float*)d_in[30];
  A.M = M; A.out = (float*)d_out;

  scan_kernel<<<dim3(N_ / 128), 256, 0, stream>>>(A);
}
